// Llama4Experts_25245817766056
// MI455X (gfx1250) — compile-verified, matrix operand order checked by reading, not run on hardware
//
#include <hip/hip_runtime.h>
#include <hip/hip_bf16.h>
#include <cstdint>

// Problem constants (match reference)
#define NE 8
#define NH 2048
#define NI 4096
#define NT 1024            // tokens per expert = 8192 / 8

typedef __attribute__((ext_vector_type(16))) __bf16          bf16x16;
typedef __attribute__((ext_vector_type(8)))  float           f32x8;
typedef __attribute__((ext_vector_type(4)))  float           f32x4;
typedef __attribute__((ext_vector_type(16))) unsigned short  u16x16;
typedef __attribute__((ext_vector_type(8)))  unsigned short  u16x8;
typedef __attribute__((ext_vector_type(4)))  unsigned int    u32x4;
typedef __attribute__((ext_vector_type(8)))  int             i32x8;
typedef __attribute__((ext_vector_type(4)))  int             i32x4;

#if defined(__has_builtin)
#  if __has_builtin(__builtin_amdgcn_tensor_load_to_lds) && __has_builtin(__builtin_amdgcn_s_wait_tensorcnt)
#    define HAVE_TDM 1
#  endif
#endif

union FragU { struct { u16x8 lo, hi; } p; u16x16 v; };

__device__ __forceinline__ unsigned short f32_to_bf16_rne(float f) {
    unsigned u = __builtin_bit_cast(unsigned, f);
    u += 0x7FFFu + ((u >> 16) & 1u);
    return (unsigned short)(u >> 16);
}

__device__ __forceinline__ f32x8 wmma_bf16(u16x16 a, u16x16 b, f32x8 c) {
    return __builtin_amdgcn_wmma_f32_16x16x32_bf16(
        false, __builtin_bit_cast(bf16x16, a),
        false, __builtin_bit_cast(bf16x16, b),
        (short)0, c, false, false);
}

// LDS tile stored [row_of_16dim][32 k] bf16, k-contiguous.
// CDNA5 16-bit operand layout: lanes 0-15 hold K{0..7,16..23}, lanes 16-31 K{8..15,24..31}.
__device__ __forceinline__ u16x16 load_frag(const unsigned short* tile, int row, int ko) {
    FragU f;
    f.p.lo = *(const u16x8*)(tile + row * 32 + ko);
    f.p.hi = *(const u16x8*)(tile + row * 32 + ko + 16);
    return f.v;
}

__device__ __forceinline__ float silu_mul(float g, float u) {
    return u * g * __builtin_amdgcn_rcpf(1.0f + __expf(-g));
}

#if defined(HAVE_TDM)
typedef __attribute__((address_space(3))) unsigned char lds_byte_t;
__device__ __forceinline__ unsigned lds_off_of(const void* p) {
    return (unsigned)(unsigned long long)(const lds_byte_t*)p;
}
// D# group 0: count=1, lds_addr, global_addr (bytes), type=2
__device__ __forceinline__ u32x4 tdm_g0(unsigned lds_off, unsigned long long ga) {
    u32x4 g;
    g[0] = 1u;
    g[1] = lds_off;
    g[2] = (unsigned)(ga & 0xFFFFFFFFu);
    g[3] = (unsigned)((ga >> 32) & 0x01FFFFFFu) | (2u << 30);
    return g;
}
// D# group 1: data_size=1 (2B), 2D tile (tile1 rows x tile0 k-elems), row stride s0.
__device__ __forceinline__ i32x8 tdm_g1(unsigned td0, unsigned td1,
                                        unsigned tile0, unsigned tile1,
                                        unsigned long long s0) {
    i32x8 g;
    g[0] = (int)(1u << 16);                                   // data_size=2B
    g[1] = (int)((td0 & 0xFFFFu) << 16);                      // tensor_dim0[15:0]
    g[2] = (int)((td0 >> 16) | ((td1 & 0xFFFFu) << 16));      // td0[31:16] | td1[15:0]
    g[3] = (int)((td1 >> 16) | (tile0 << 16));                // td1[31:16] | tile_dim0
    g[4] = (int)(tile1 & 0xFFFFu);                            // tile_dim1 (tile_dim2=0)
    g[5] = (int)(unsigned)(s0 & 0xFFFFFFFFu);                 // stride0[31:0]
    g[6] = (int)(unsigned)((s0 >> 32) & 0xFFFFu);             // stride0[47:32]
    g[7] = 0;
    return g;
}
__device__ __forceinline__ void tdm_load(u32x4 g0, i32x8 g1) {
    i32x4 z4 = {0, 0, 0, 0};
#  if __clang_major__ >= 23
    i32x8 z8 = {0, 0, 0, 0, 0, 0, 0, 0};
    __builtin_amdgcn_tensor_load_to_lds(g0, g1, z4, z4, z8, 0);
#  else
    __builtin_amdgcn_tensor_load_to_lds(g0, g1, z4, z4, 0);
#  endif
}
#endif // HAVE_TDM

// ---------------------------------------------------------------------------
// Preprocess 1: elementwise fp32 -> bf16 (hidden states).
// ---------------------------------------------------------------------------
__global__ __launch_bounds__(256)
void cvt_kernel(const float* __restrict__ src, unsigned short* __restrict__ dst) {
    size_t i = ((size_t)blockIdx.x * 256 + threadIdx.x) * 8;
    f32x4 a = *(const f32x4*)(src + i);
    f32x4 b = *(const f32x4*)(src + i + 4);
    u16x8 o;
#pragma unroll
    for (int j = 0; j < 4; ++j) { o[j] = f32_to_bf16_rne(a[j]); o[4 + j] = f32_to_bf16_rne(b[j]); }
    *(u16x8*)(dst + i) = o;
}

// ---------------------------------------------------------------------------
// Preprocess 2: tiled transpose + convert. src [E][R][C] fp32 -> dst [E][C][R] bf16.
// ---------------------------------------------------------------------------
__global__ __launch_bounds__(256)
void tconv_kernel(const float* __restrict__ src, unsigned short* __restrict__ dst,
                  int R, int C) {
    const int e  = blockIdx.z;
    const int r0 = blockIdx.y * 64;
    const int c0 = blockIdx.x * 64;
    const float*    S = src + (size_t)e * R * C;
    unsigned short* D = dst + (size_t)e * R * C;

    __shared__ unsigned short t[64][64 + 2];

    const int tid = threadIdx.x;
    {
        const int row = tid >> 2;
        const int cc  = (tid & 3) * 16;
        const float* s = S + (size_t)(r0 + row) * C + c0 + cc;
#pragma unroll
        for (int j = 0; j < 4; ++j) {
            f32x4 v = *(const f32x4*)(s + j * 4);
#pragma unroll
            for (int i = 0; i < 4; ++i) t[row][cc + j * 4 + i] = f32_to_bf16_rne(v[i]);
        }
    }
    __syncthreads();
    {
        const int c  = tid >> 2;
        const int rr = (tid & 3) * 16;
        unsigned short* d = D + (size_t)(c0 + c) * R + r0 + rr;
        u16x8 o0, o1;
#pragma unroll
        for (int i = 0; i < 8; ++i) { o0[i] = t[rr + i][c]; o1[i] = t[rr + 8 + i][c]; }
        *(u16x8*)(d)     = o0;
        *(u16x8*)(d + 8) = o1;
    }
}

// ---------------------------------------------------------------------------
// GEMM1: act = SwiGLU(X @ W1). bf16 operands, TDM-staged double-buffered LDS.
// C tile 128x64, 8 waves (4Mx2N), wave tile 32x32, 8 WMMA/wave/k-step.
// grid: (NI/64, NT/128, NE)
// ---------------------------------------------------------------------------
__global__ __launch_bounds__(256)
void gemm1_kernel(const unsigned short* __restrict__ Xb,
                  const unsigned short* __restrict__ W1T,
                  unsigned short* __restrict__ ACT) {
    const int e  = blockIdx.z;
    const int m0 = blockIdx.y * 128;
    const int n0 = blockIdx.x * 64;

    __shared__ unsigned short sA [2][128 * 32];
    __shared__ unsigned short sBg[2][64 * 32];
    __shared__ unsigned short sBu[2][64 * 32];

    const int tid  = threadIdx.x;
    const int wave = tid >> 5;
    const int lane = tid & 31;
    const int hl   = lane >> 4;
    const int lr   = lane & 15;
    const int wm   = (wave >> 1) * 32;
    const int wn   = (wave & 1) * 32;

    const unsigned short* Ab = Xb  + ((size_t)e * NT + m0) * NH;
    const unsigned short* Gb = W1T + (size_t)e * (2 * NI) * NH + (size_t)n0 * NH;
    const unsigned short* Ub = Gb + (size_t)NI * NH;

    f32x8 ag[2][2] = {}, au[2][2] = {};
    const int NK = NH / 32;

#if defined(HAVE_TDM)
    const i32x8 dA = tdm_g1(32, 128, 32, 128, NH);
    const i32x8 dB = tdm_g1(32, 64,  32, 64,  NH);
    if (wave == 0) {
        tdm_load(tdm_g0(lds_off_of(&sA [0][0]), (unsigned long long)(uintptr_t)Ab), dA);
        tdm_load(tdm_g0(lds_off_of(&sBg[0][0]), (unsigned long long)(uintptr_t)Gb), dB);
        tdm_load(tdm_g0(lds_off_of(&sBu[0][0]), (unsigned long long)(uintptr_t)Ub), dB);
    }
    for (int k = 0; k < NK; ++k) {
        const int cur = k & 1, nxt = cur ^ 1;
        __syncthreads();   // previous compute done reading buf 'nxt'
        if (wave == 0) {
            if (k + 1 < NK) {
                const size_t kb = (size_t)(k + 1) * 32 * 2;   // byte offset along k
                tdm_load(tdm_g0(lds_off_of(&sA [nxt][0]), (unsigned long long)(uintptr_t)Ab + kb), dA);
                tdm_load(tdm_g0(lds_off_of(&sBg[nxt][0]), (unsigned long long)(uintptr_t)Gb + kb), dB);
                tdm_load(tdm_g0(lds_off_of(&sBu[nxt][0]), (unsigned long long)(uintptr_t)Ub + kb), dB);
                __builtin_amdgcn_s_wait_tensorcnt(3);  // tile k complete, k+1 in flight
            } else {
                __builtin_amdgcn_s_wait_tensorcnt(0);
            }
        }
        __syncthreads();   // buf 'cur' published
#else
    const int ar = tid >> 1, ak = (tid & 1) * 16;
    const int br = tid >> 2, bk = (tid & 3) * 8;
    u16x8 ra0 = *(const u16x8*)(Ab + (size_t)ar * NH + ak);
    u16x8 ra1 = *(const u16x8*)(Ab + (size_t)ar * NH + ak + 8);
    u16x8 rg  = *(const u16x8*)(Gb + (size_t)br * NH + bk);
    u16x8 ru  = *(const u16x8*)(Ub + (size_t)br * NH + bk);
    *(u16x8*)(&sA[0][ar * 32 + ak])     = ra0;
    *(u16x8*)(&sA[0][ar * 32 + ak + 8]) = ra1;
    *(u16x8*)(&sBg[0][br * 32 + bk])    = rg;
    *(u16x8*)(&sBu[0][br * 32 + bk])    = ru;
    for (int k = 0; k < NK; ++k) {
        const int cur = k & 1, nxt = cur ^ 1;
        __syncthreads();
        const bool more = (k + 1) < NK;
        if (more) {
            const int ko2 = (k + 1) * 32;
            ra0 = *(const u16x8*)(Ab + (size_t)ar * NH + ko2 + ak);
            ra1 = *(const u16x8*)(Ab + (size_t)ar * NH + ko2 + ak + 8);
            rg  = *(const u16x8*)(Gb + (size_t)br * NH + ko2 + bk);
            ru  = *(const u16x8*)(Ub + (size_t)br * NH + ko2 + bk);
        }
#endif
        const int ko = hl * 8;
        u16x16 a0  = load_frag(sA[cur],  wm + lr,      ko);
        u16x16 a1  = load_frag(sA[cur],  wm + 16 + lr, ko);
        u16x16 bg0 = load_frag(sBg[cur], wn + lr,      ko);
        u16x16 bg1 = load_frag(sBg[cur], wn + 16 + lr, ko);
        u16x16 bu0 = load_frag(sBu[cur], wn + lr,      ko);
        u16x16 bu1 = load_frag(sBu[cur], wn + 16 + lr, ko);

        ag[0][0] = wmma_bf16(a0, bg0, ag[0][0]);
        ag[0][1] = wmma_bf16(a0, bg1, ag[0][1]);
        ag[1][0] = wmma_bf16(a1, bg0, ag[1][0]);
        ag[1][1] = wmma_bf16(a1, bg1, ag[1][1]);
        au[0][0] = wmma_bf16(a0, bu0, au[0][0]);
        au[0][1] = wmma_bf16(a0, bu1, au[0][1]);
        au[1][0] = wmma_bf16(a1, bu0, au[1][0]);
        au[1][1] = wmma_bf16(a1, bu1, au[1][1]);

#if !defined(HAVE_TDM)
        if (more) {
            *(u16x8*)(&sA[nxt][ar * 32 + ak])     = ra0;
            *(u16x8*)(&sA[nxt][ar * 32 + ak + 8]) = ra1;
            *(u16x8*)(&sBg[nxt][br * 32 + bk])    = rg;
            *(u16x8*)(&sBu[nxt][br * 32 + bk])    = ru;
        }
#else
        (void)nxt;
#endif
    }

    unsigned short* Ap = ACT + ((size_t)e * NT + m0) * NI + n0;
#pragma unroll
    for (int ms = 0; ms < 2; ++ms) {
#pragma unroll
        for (int r = 0; r < 8; ++r) {
            const int m = wm + ms * 16 + hl * 8 + r;
#pragma unroll
            for (int ns = 0; ns < 2; ++ns) {
                float s = silu_mul(ag[ms][ns][r], au[ms][ns][r]);
                Ap[(size_t)m * NI + wn + ns * 16 + lr] = f32_to_bf16_rne(s);
            }
        }
    }
}

// ---------------------------------------------------------------------------
// GEMM2: out = act @ W2. C tile 128x64, 4 WMMA/wave/k-step.
// grid: (NH/64, NT/128, NE)
// ---------------------------------------------------------------------------
__global__ __launch_bounds__(256)
void gemm2_kernel(const unsigned short* __restrict__ ACT,
                  const unsigned short* __restrict__ W2T,
                  float* __restrict__ OUT) {
    const int e  = blockIdx.z;
    const int m0 = blockIdx.y * 128;
    const int n0 = blockIdx.x * 64;

    __shared__ unsigned short sA[2][128 * 32];
    __shared__ unsigned short sB[2][64 * 32];

    const int tid  = threadIdx.x;
    const int wave = tid >> 5;
    const int lane = tid & 31;
    const int hl   = lane >> 4;
    const int lr   = lane & 15;
    const int wm   = (wave >> 1) * 32;
    const int wn   = (wave & 1) * 32;

    const unsigned short* Ab = ACT + ((size_t)e * NT + m0) * NI;
    const unsigned short* Bb = W2T + (size_t)e * NH * NI + (size_t)n0 * NI;

    f32x8 acc[2][2] = {};
    const int NK = NI / 32;

#if defined(HAVE_TDM)
    const i32x8 dA = tdm_g1(32, 128, 32, 128, NI);
    const i32x8 dB = tdm_g1(32, 64,  32, 64,  NI);
    if (wave == 0) {
        tdm_load(tdm_g0(lds_off_of(&sA[0][0]), (unsigned long long)(uintptr_t)Ab), dA);
        tdm_load(tdm_g0(lds_off_of(&sB[0][0]), (unsigned long long)(uintptr_t)Bb), dB);
    }
    for (int k = 0; k < NK; ++k) {
        const int cur = k & 1, nxt = cur ^ 1;
        __syncthreads();
        if (wave == 0) {
            if (k + 1 < NK) {
                const size_t kb = (size_t)(k + 1) * 32 * 2;
                tdm_load(tdm_g0(lds_off_of(&sA[nxt][0]), (unsigned long long)(uintptr_t)Ab + kb), dA);
                tdm_load(tdm_g0(lds_off_of(&sB[nxt][0]), (unsigned long long)(uintptr_t)Bb + kb), dB);
                __builtin_amdgcn_s_wait_tensorcnt(2);
            } else {
                __builtin_amdgcn_s_wait_tensorcnt(0);
            }
        }
        __syncthreads();
#else
    const int ar = tid >> 1, ak = (tid & 1) * 16;
    const int br = tid >> 2, bk = (tid & 3) * 8;
    u16x8 ra0 = *(const u16x8*)(Ab + (size_t)ar * NI + ak);
    u16x8 ra1 = *(const u16x8*)(Ab + (size_t)ar * NI + ak + 8);
    u16x8 rb  = *(const u16x8*)(Bb + (size_t)br * NI + bk);
    *(u16x8*)(&sA[0][ar * 32 + ak])     = ra0;
    *(u16x8*)(&sA[0][ar * 32 + ak + 8]) = ra1;
    *(u16x8*)(&sB[0][br * 32 + bk])     = rb;
    for (int k = 0; k < NK; ++k) {
        const int cur = k & 1, nxt = cur ^ 1;
        __syncthreads();
        const bool more = (k + 1) < NK;
        if (more) {
            const int ko2 = (k + 1) * 32;
            ra0 = *(const u16x8*)(Ab + (size_t)ar * NI + ko2 + ak);
            ra1 = *(const u16x8*)(Ab + (size_t)ar * NI + ko2 + ak + 8);
            rb  = *(const u16x8*)(Bb + (size_t)br * NI + ko2 + bk);
        }
#endif
        const int ko = hl * 8;
        u16x16 a0 = load_frag(sA[cur], wm + lr,      ko);
        u16x16 a1 = load_frag(sA[cur], wm + 16 + lr, ko);
        u16x16 b0 = load_frag(sB[cur], wn + lr,      ko);
        u16x16 b1 = load_frag(sB[cur], wn + 16 + lr, ko);

        acc[0][0] = wmma_bf16(a0, b0, acc[0][0]);
        acc[0][1] = wmma_bf16(a0, b1, acc[0][1]);
        acc[1][0] = wmma_bf16(a1, b0, acc[1][0]);
        acc[1][1] = wmma_bf16(a1, b1, acc[1][1]);

#if !defined(HAVE_TDM)
        if (more) {
            *(u16x8*)(&sA[nxt][ar * 32 + ak])     = ra0;
            *(u16x8*)(&sA[nxt][ar * 32 + ak + 8]) = ra1;
            *(u16x8*)(&sB[nxt][br * 32 + bk])     = rb;
        }
#else
        (void)nxt;
#endif
    }

    float* Op = OUT + ((size_t)e * NT + m0) * NH + n0;
#pragma unroll
    for (int ms = 0; ms < 2; ++ms) {
#pragma unroll
        for (int r = 0; r < 8; ++r) {
            const int m = wm + ms * 16 + hl * 8 + r;
#pragma unroll
            for (int ns = 0; ns < 2; ++ns) {
                Op[(size_t)m * NH + wn + ns * 16 + lr] = acc[ms][ns][r];
            }
        }
    }
}

// ---------------------------------------------------------------------------
extern "C" void kernel_launch(void* const* d_in, const int* in_sizes, int n_in,
                              void* d_out, int out_size, void* d_ws, size_t ws_size,
                              hipStream_t stream) {
    const float* hidden  = (const float*)d_in[0];   // [8192, 2048]
    const float* gate_up = (const float*)d_in[1];   // [8, 2048, 8192]
    const float* down    = (const float*)d_in[2];   // [8, 4096, 2048]
    float*       out     = (float*)d_out;           // [8192, 2048]

    // Workspace (bf16): Xb 33.6MB | W1T 268.4MB | W2T 134.2MB | ACT 67.1MB
    unsigned short* Xb  = (unsigned short*)d_ws;
    unsigned short* W1T = Xb  + (size_t)NE * NT * NH;
    unsigned short* W2T = W1T + (size_t)NE * (2 * NI) * NH;
    unsigned short* ACT = W2T + (size_t)NE * NH * NI;

    dim3 block(256, 1, 1);

    cvt_kernel<<<dim3((NE * NT * NH) / (256 * 8), 1, 1), block, 0, stream>>>(hidden, Xb);
    tconv_kernel<<<dim3((2 * NI) / 64, NH / 64, NE), block, 0, stream>>>(gate_up, W1T, NH, 2 * NI);
    tconv_kernel<<<dim3(NH / 64, NI / 64, NE), block, 0, stream>>>(down, W2T, NI, NH);

    gemm1_kernel<<<dim3(NI / 64, NT / 128, NE), block, 0, stream>>>(Xb, W1T, ACT);
    gemm2_kernel<<<dim3(NH / 64, NT / 128, NE), block, 0, stream>>>(ACT, W2T, out);
}
